// MemoryModule_72112500900108
// MI455X (gfx1250) — matrix operand order repaired and checked by reading, not compiled, not run on hardware
//
#include <hip/hip_runtime.h>

typedef __attribute__((ext_vector_type(16))) __bf16 v16bf;
typedef __attribute__((ext_vector_type(8)))  float  v8f;
typedef __attribute__((ext_vector_type(4)))  float  v4f;   // native vector for nontemporal ops

#define MEM_DIM        128
#define ROWS_PER_BLOCK 64
#define THREADS        128

// ---------- helpers ----------
__device__ __forceinline__ unsigned short f2bf(float f) {
  unsigned u = __float_as_uint(f);
  u += 0x7FFFu + ((u >> 16) & 1u);          // round-to-nearest-even
  return (unsigned short)(u >> 16);
}
__device__ __forceinline__ unsigned int pack2bf(float lo, float hi) {
  return (unsigned int)f2bf(lo) | ((unsigned int)f2bf(hi) << 16);
}
__device__ __forceinline__ float fast_sigmoid(float x) {
  return 1.0f / (1.0f + __expf(-x));
}
__device__ __forceinline__ float fast_tanh(float x) {
  x = fminf(fmaxf(x, -15.0f), 15.0f);
  float e = __expf(2.0f * x);
  return (e - 1.0f) / (e + 1.0f);
}

// ---------- kernel 0: fp32 weights -> bf16 fragment-major layout ----------
// Output: 192 fragments of 1 KB. Fragment id = ((c*4 + k)*3 + g)*2 + mat,
// inside a fragment lane L owns 16 bf16 = W[mat][g*128 + c*16 + (L%16)]
//                                          [k*32 + (L/16)*16 .. +15]
// so a WMMA B-fragment load is two fully coalesced 512B requests and one
// c-tile's 24 fragments are a contiguous 24 KB stream.
__global__ void wconvert_kernel(const float* __restrict__ wih,
                                const float* __restrict__ whh,
                                unsigned short* __restrict__ dst) {
  int t = blockIdx.x * blockDim.x + threadIdx.x;   // one thread = 8 bf16 outputs
  if (t >= (2 * 3 * MEM_DIM * MEM_DIM) / 8) return; // 12288 threads
  int frag   = t >> 6;          // 64 groups of 8 per 512-element fragment
  int within = t & 63;
  int lane   = within >> 1;     // 0..31
  int ehalf  = within & 1;      // low/high 8 of the lane's 16 values

  int mat  = frag & 1;
  int rest = frag >> 1;         // (c*4 + k)*3 + g
  int g    = rest % 3;
  int ck   = rest / 3;
  int k    = ck & 3;
  int c    = ck >> 2;

  int row = g * MEM_DIM + c * 16 + (lane & 15);
  int col = k * 32 + (lane >> 4) * 16 + ehalf * 8;

  const float* src = (mat ? whh : wih) + (size_t)row * MEM_DIM + col;
  float4 a = *(const float4*)(src);
  float4 b = *(const float4*)(src + 4);
  uint4 o;
  o.x = pack2bf(a.x, a.y);
  o.y = pack2bf(a.z, a.w);
  o.z = pack2bf(b.x, b.y);
  o.w = pack2bf(b.z, b.w);
  ((uint4*)dst)[t] = o;
}

// ---------- kernel 1: streaming copy memory -> out (512 MB each way) ----------
__global__ void memcopy4_kernel(const v4f* __restrict__ src,
                                v4f* __restrict__ dst, int n4) {
  int i = blockIdx.x * blockDim.x + threadIdx.x;
  int stride = gridDim.x * blockDim.x;
  for (; i < n4; i += stride) {
    v4f v = __builtin_nontemporal_load(src + i);
    __builtin_nontemporal_store(v, dst + i);
  }
}

// ---------- kernel 2: gather -> GRU (bf16 WMMA) -> scatter ----------
__global__ void __launch_bounds__(THREADS)
gru_scatter_kernel(const int* __restrict__ node_ids,
                   const float* __restrict__ messages,
                   const float* __restrict__ memory,
                   const unsigned short* __restrict__ wbf,   // fragment-major bf16 weights
                   const float* __restrict__ b_ih,
                   const float* __restrict__ b_hh,
                   float* __restrict__ out) {
  __shared__ unsigned short s_x[ROWS_PER_BLOCK * MEM_DIM];  // 16 KB bf16 messages tile
  __shared__ float          s_h[ROWS_PER_BLOCK * MEM_DIM];  // 32 KB f32 gathered memory
  __shared__ float          s_bih[3 * MEM_DIM];
  __shared__ float          s_bhh[3 * MEM_DIM];
  __shared__ int            s_nid[ROWS_PER_BLOCK];

  const int tid  = threadIdx.x;
  const int base = blockIdx.x * ROWS_PER_BLOCK;

  if (tid < ROWS_PER_BLOCK) s_nid[tid] = node_ids[base + tid];
  for (int i = tid; i < 3 * MEM_DIM; i += THREADS) {
    s_bih[i] = b_ih[i];
    s_bhh[i] = b_hh[i];
  }
  __syncthreads();

  // Stage x (convert to bf16) and gathered h rows (keep fp32 for the blend).
  const int ITERS = (ROWS_PER_BLOCK * MEM_DIM / 4) / THREADS;  // 16
  for (int j = 0; j < ITERS; ++j) {
    int flat = j * THREADS + tid;               // float4 index within tile
    int row  = flat >> 5;                       // 32 float4 per row
    int c4   = flat & 31;
    float4 xv = ((const float4*)messages)[(size_t)(base + row) * 32 + c4];
    uint2 xp;
    xp.x = pack2bf(xv.x, xv.y);
    xp.y = pack2bf(xv.z, xv.w);
    ((uint2*)s_x)[flat] = xp;
    float4 hv = ((const float4*)memory)[(size_t)s_nid[row] * 32 + c4];
    ((float4*)s_h)[flat] = hv;
  }
  __syncthreads();

  const int lane = tid & 31;
  const int wave = tid >> 5;
  const int m    = lane & 15;     // A-matrix row within wave tile (lanes 0-15 / 16-31 both M=0..15)
  const int kh   = lane >> 4;     // K half-select
  const int rowbase = wave * 16 + m;

  // Build A fragments once per K-chunk (16x32 bf16 each, 4 chunks cover K=128).
  // Lane L holds row M=L%16, K = {kc*32 + kh*16 .. +15}.
  v16bf ax[4], ah[4];
#pragma unroll
  for (int k = 0; k < 4; ++k) {
    const int off = rowbase * MEM_DIM + k * 32 + kh * 16;
    union { uint4 q[2]; v16bf v; } ua;
    ua.q[0] = *(const uint4*)(s_x + off);
    ua.q[1] = *(const uint4*)(s_x + off + 8);
    ax[k] = ua.v;
    const float* ph = s_h + off;
    float4 f0 = *(const float4*)(ph + 0);
    float4 f1 = *(const float4*)(ph + 4);
    float4 f2 = *(const float4*)(ph + 8);
    float4 f3 = *(const float4*)(ph + 12);
    union { unsigned int u[8]; v16bf v; } uh;
    uh.u[0] = pack2bf(f0.x, f0.y); uh.u[1] = pack2bf(f0.z, f0.w);
    uh.u[2] = pack2bf(f1.x, f1.y); uh.u[3] = pack2bf(f1.z, f1.w);
    uh.u[4] = pack2bf(f2.x, f2.y); uh.u[5] = pack2bf(f2.z, f2.w);
    uh.u[6] = pack2bf(f3.x, f3.y); uh.u[7] = pack2bf(f3.z, f3.w);
    ah[k] = uh.v;
  }

  const int n = lane & 15;          // B/C column within tile

  // 8 column tiles of 16 cover the 128 hidden dims; gates r/z/n live at
  // column offsets 0/128/256 of the 384-wide GEMM output.
  for (int c = 0; c < MEM_DIM / 16; ++c) {
    v8f acc_ir = {}, acc_iz = {}, acc_in = {};
    v8f acc_hr = {}, acc_hz = {}, acc_hn = {};
    // base of this c-tile's contiguous 24 KB fragment stream
    const unsigned short* cbase = wbf + (size_t)(c * 4 * 3 * 2) * 512 + lane * 16;
#pragma unroll
    for (int k = 0; k < 4; ++k) {
#pragma unroll
      for (int g = 0; g < 3; ++g) {
        const unsigned short* pwi = cbase + (size_t)((k * 3 + g) * 2) * 512;
        const unsigned short* pwh = pwi + 512;
        union { uint4 q[2]; v16bf v; } bi, bh;
        bi.q[0] = *(const uint4*)(pwi);
        bi.q[1] = *(const uint4*)(pwi + 8);
        bh.q[0] = *(const uint4*)(pwh);
        bh.q[1] = *(const uint4*)(pwh + 8);
        v8f* acci = (g == 0) ? &acc_ir : (g == 1) ? &acc_iz : &acc_in;
        v8f* acch = (g == 0) ? &acc_hr : (g == 1) ? &acc_hz : &acc_hn;
        *acci = __builtin_amdgcn_wmma_f32_16x16x32_bf16(
            false, ax[k], false, bi.v, (short)0, *acci, false, false);
        *acch = __builtin_amdgcn_wmma_f32_16x16x32_bf16(
            false, ah[k], false, bh.v, (short)0, *acch, false, false);
      }
    }

    // Gate math directly on accumulator layout:
    // element (VGPR j, lane L): row = j + 8*(L>=16), col = c*16 + L%16
    const int col = c * 16 + n;
    const float bir = s_bih[col], biz = s_bih[MEM_DIM + col], bin = s_bih[2 * MEM_DIM + col];
    const float bhr = s_bhh[col], bhz = s_bhh[MEM_DIM + col], bhn = s_bhh[2 * MEM_DIM + col];
#pragma unroll
    for (int j = 0; j < 8; ++j) {
      const int grow = wave * 16 + kh * 8 + j;
      float r  = fast_sigmoid(acc_ir[j] + bir + acc_hr[j] + bhr);
      float z  = fast_sigmoid(acc_iz[j] + biz + acc_hz[j] + bhz);
      float nn = fast_tanh(acc_in[j] + bin + r * (acc_hn[j] + bhn));
      float h  = s_h[grow * MEM_DIM + col];
      float o  = (1.0f - z) * nn + z * h;
      out[(size_t)s_nid[grow] * MEM_DIM + col] = o;
    }
  }
}

// ---------- host entry ----------
extern "C" void kernel_launch(void* const* d_in, const int* in_sizes, int n_in,
                              void* d_out, int out_size, void* d_ws, size_t ws_size,
                              hipStream_t stream) {
  const int*   node_ids = (const int*)d_in[0];
  const float* messages = (const float*)d_in[1];
  const float* memory   = (const float*)d_in[2];
  const float* W_ih     = (const float*)d_in[3];
  const float* W_hh     = (const float*)d_in[4];
  const float* b_ih     = (const float*)d_in[5];
  const float* b_hh     = (const float*)d_in[6];
  float* out = (float*)d_out;
  unsigned short* wbf = (unsigned short*)d_ws;   // 192 KB fragment-major bf16 weights

  // 1) weights fp32 -> bf16, swizzled into WMMA fragment-major order
  wconvert_kernel<<<48, 256, 0, stream>>>(W_ih, W_hh, wbf);

  // 2) bulk copy memory -> out (the bandwidth floor of this problem)
  const int n4 = out_size / 4;                   // 32M float4
  memcopy4_kernel<<<8192, 256, 0, stream>>>((const v4f*)memory, (v4f*)out, n4);

  // 3) gather + GRU (bf16 WMMA) + scatter
  const int batch  = in_sizes[0];
  const int blocks = batch / ROWS_PER_BLOCK;     // 200000/64 = 3125 exact
  gru_scatter_kernel<<<blocks, THREADS, 0, stream>>>(
      node_ids, messages, memory, wbf, b_ih, b_hh, out);
}
